// GPT_84954453115587
// MI455X (gfx1250) — matrix operand order repaired and checked by reading, not compiled, not run on hardware
//
#include <hip/hip_runtime.h>
#include <hip/hip_bf16.h>
#include <math.h>

typedef __attribute__((ext_vector_type(16))) __bf16 v16bf;
typedef __attribute__((ext_vector_type(8)))  float  v8f;
typedef unsigned int u32x4 __attribute__((ext_vector_type(4)));
typedef unsigned int u32x8 __attribute__((ext_vector_type(8)));

#define DEVFN static __device__ __forceinline__

DEVFN unsigned short f2bf(float f) {
    unsigned int u = __float_as_uint(f);
    return (unsigned short)((u + 0x7FFFu + ((u >> 16) & 1u)) >> 16);
}

// ---------------- elementwise f32 -> bf16 ----------------
__global__ void k_f32_to_bf16(const float* __restrict__ src,
                              unsigned short* __restrict__ dst, long long n) {
    long long i = (long long)blockIdx.x * blockDim.x + threadIdx.x;
    long long stride = (long long)gridDim.x * blockDim.x;
    for (; i < n; i += stride) dst[i] = f2bf(src[i]);
}

// ---------------- token + position embedding ----------------
__global__ void k_embed(const int* __restrict__ x, const float* __restrict__ wte,
                        const float* __restrict__ wpe, float* __restrict__ h) {
    int row = blockIdx.x;          // b*1024 + t
    int t = row & 1023;
    int tok = x[row];
    const float* we = wte + (size_t)tok * 768;
    const float* wp = wpe + (size_t)t * 768;
    float* out = h + (size_t)row * 768;
    for (int e = threadIdx.x; e < 768; e += blockDim.x) out[e] = we[e] + wp[e];
}

// ---------------- layernorm (f32 in, bf16 out) ----------------
__global__ void k_layernorm(const float* __restrict__ x, const float* __restrict__ w,
                            const float* __restrict__ b, unsigned short* __restrict__ out) {
    int row = blockIdx.x;
    const float* xr = x + (size_t)row * 768;
    float s = 0.f, s2 = 0.f;
    for (int e = threadIdx.x; e < 768; e += blockDim.x) { float v = xr[e]; s += v; s2 += v * v; }
    __shared__ float sh[16];
    int lane = threadIdx.x & 31, wave = threadIdx.x >> 5;
    for (int o = 16; o > 0; o >>= 1) { s += __shfl_xor(s, o, 32); s2 += __shfl_xor(s2, o, 32); }
    if (lane == 0) { sh[wave] = s; sh[8 + wave] = s2; }
    __syncthreads();
    if (wave == 0) {
        float a  = (lane < 8) ? sh[lane] : 0.f;
        float a2 = (lane < 8) ? sh[8 + lane] : 0.f;
        for (int o = 16; o > 0; o >>= 1) { a += __shfl_xor(a, o, 32); a2 += __shfl_xor(a2, o, 32); }
        if (lane == 0) { sh[0] = a; sh[8] = a2; }
    }
    __syncthreads();
    float mean = sh[0] * (1.f / 768.f);
    float var  = sh[8] * (1.f / 768.f) - mean * mean;
    float rstd = rsqrtf(var + 1e-5f);
    unsigned short* orow = out + (size_t)row * 768;
    for (int e = threadIdx.x; e < 768; e += blockDim.x)
        orow[e] = f2bf((xr[e] - mean) * rstd * w[e] + b[e]);
}

// ---------------- qkv [B,T,3E] f32 -> q/k/v [B,H,T,D] bf16 ----------------
__global__ void k_pack_qkv(const float* __restrict__ qkv, unsigned short* __restrict__ q,
                           unsigned short* __restrict__ k, unsigned short* __restrict__ v) {
    long long i = (long long)blockIdx.x * blockDim.x + threadIdx.x;
    if (i >= 2LL * 1024 * 768) return;
    int e = (int)(i % 768);
    long long bt = i / 768;
    int b = (int)(bt / 1024), t = (int)(bt % 1024);
    int hh = e >> 6, d = e & 63;
    size_t dst = ((size_t)(b * 12 + hh) * 1024 + t) * 64 + d;
    const float* src = qkv + bt * 2304;
    q[dst] = f2bf(src[e]);
    k[dst] = f2bf(src[768 + e]);
    v[dst] = f2bf(src[1536 + e]);
}

// ---------------- Y [B,H,T,D] f32 -> y [B,T,E] bf16 ----------------
__global__ void k_pack_y(const float* __restrict__ Y, unsigned short* __restrict__ yb) {
    long long i = (long long)blockIdx.x * blockDim.x + threadIdx.x;
    if (i >= 2LL * 1024 * 768) return;
    int e = (int)(i % 768);
    long long bt = i / 768;
    int b = (int)(bt / 1024), t = (int)(bt % 1024);
    int hh = e >> 6, d = e & 63;
    yb[i] = f2bf(Y[((size_t)(b * 12 + hh) * 1024 + t) * 64 + d]);
}

// ---------------- row softmax (1024 cols, f32 in, bf16 out) ----------------
__global__ void k_softmax_row(const float* __restrict__ S, unsigned short* __restrict__ P) {
    size_t row = blockIdx.x;
    const float* sr = S + row * 1024;
    unsigned short* pr = P + row * 1024;
    int lane = threadIdx.x & 31, wave = threadIdx.x >> 5;
    __shared__ float sh[8];
    float mx = -3.0e38f;
    for (int j = threadIdx.x; j < 1024; j += 256) mx = fmaxf(mx, sr[j]);
    for (int o = 16; o > 0; o >>= 1) mx = fmaxf(mx, __shfl_xor(mx, o, 32));
    if (lane == 0) sh[wave] = mx;
    __syncthreads();
    if (wave == 0) {
        float a = (lane < 8) ? sh[lane] : -3.0e38f;
        for (int o = 16; o > 0; o >>= 1) a = fmaxf(a, __shfl_xor(a, o, 32));
        if (lane == 0) sh[0] = a;
    }
    __syncthreads();
    mx = sh[0];
    __syncthreads();
    float sum = 0.f;
    for (int j = threadIdx.x; j < 1024; j += 256) sum += __expf(sr[j] - mx);
    for (int o = 16; o > 0; o >>= 1) sum += __shfl_xor(sum, o, 32);
    if (lane == 0) sh[wave] = sum;
    __syncthreads();
    if (wave == 0) {
        float a = (lane < 8) ? sh[lane] : 0.f;
        for (int o = 16; o > 0; o >>= 1) a += __shfl_xor(a, o, 32);
        if (lane == 0) sh[0] = a;
    }
    __syncthreads();
    float inv = 1.f / sh[0];
    for (int j = threadIdx.x; j < 1024; j += 256) pr[j] = f2bf(__expf(sr[j] - mx) * inv);
}

// ---------------- cross-entropy loss over logits rows ----------------
__global__ void k_zero(float* p) { if (threadIdx.x == 0 && blockIdx.x == 0) *p = 0.f; }

__global__ void k_loss(const float* __restrict__ logits, const int* __restrict__ label,
                       float* __restrict__ loss) {
    int row = blockIdx.x;
    const float* lr = logits + (size_t)row * 50257;
    int lane = threadIdx.x & 31, wave = threadIdx.x >> 5;
    __shared__ float sh[8];
    float mx = -3.0e38f;
    for (int j = threadIdx.x; j < 50257; j += 256) mx = fmaxf(mx, lr[j]);
    for (int o = 16; o > 0; o >>= 1) mx = fmaxf(mx, __shfl_xor(mx, o, 32));
    if (lane == 0) sh[wave] = mx;
    __syncthreads();
    if (wave == 0) {
        float a = (lane < 8) ? sh[lane] : -3.0e38f;
        for (int o = 16; o > 0; o >>= 1) a = fmaxf(a, __shfl_xor(a, o, 32));
        if (lane == 0) sh[0] = a;
    }
    __syncthreads();
    mx = sh[0];
    __syncthreads();
    float sum = 0.f;
    for (int j = threadIdx.x; j < 50257; j += 256) sum += __expf(lr[j] - mx);
    for (int o = 16; o > 0; o >>= 1) sum += __shfl_xor(sum, o, 32);
    if (lane == 0) sh[wave] = sum;
    __syncthreads();
    if (wave == 0) {
        float a = (lane < 8) ? sh[lane] : 0.f;
        for (int o = 16; o > 0; o >>= 1) a += __shfl_xor(a, o, 32);
        if (lane == 0) sh[0] = a;
    }
    __syncthreads();
    if (threadIdx.x == 0)
        atomicAdd(loss, (mx + logf(sh[0]) - lr[label[row]]) * (1.f / 2048.f));
}

// ---------------- universal WMMA GEMM ----------------
// C[z] = A[z][M,K](bf16) * B[z](bf16) (+bias) (+residual) (gelu/causal) -> f32/bf16
// BT=false: A via async-to-LDS (ASYNCcnt), B via TDM tensor_load_to_lds (TENSORcnt),
//           double-buffered.
// BT=true:  B is [N,K] row-major (logical B[k][n]=Bm[n*K+k]); sync transpose staging.
// flags: bit0 = tanh-GELU, bit1 = causal mask + scale (attention scores).
#define FLAG_GELU   1
#define FLAG_CAUSAL 2

// CDNA5 async copy: VDST = per-lane LDS byte offset, VADDR = per-lane 64-bit
// global address (GV mode). Tracked with ASYNCcnt (CDNA5 ISA).
#define ASYNC_B128(lds32, gaddr64)                                            \
    asm volatile("global_load_async_to_lds_b128 %0, %1, off"                  \
                 :: "v"(lds32), "v"(gaddr64) : "memory")
#define WAIT_ASYNC() asm volatile("s_wait_asynccnt 0" ::: "memory")

// TDM: DMA a 2D tile (rows x cols, bf16) from a [*, rowStrideElems] tensor into
// LDS (contiguous row-major tile). D# per CDNA5 ISA.
// 2-group form (VADDR2/3 = NULL) is legal for tensors up to 2D.
DEVFN void tdm_load_tile_2d(unsigned ldsAddr, unsigned long long gByteAddr,
                            unsigned cols, unsigned rows,
                            unsigned rowStrideElems, unsigned tensorRows) {
    u32x4 g0;
    g0.x = 1u;                                          // count=1, user mode
    g0.y = ldsAddr;                                     // lds_addr [63:32]
    g0.z = (unsigned)gByteAddr;                         // global_addr [95:64]
    g0.w = (unsigned)(gByteAddr >> 32) | (2u << 30);    // addr[56:32] | type=2
    u32x8 g1;
    g1.s0 = 0x10000u;                                   // data_size=1 (2 bytes)
    g1.s1 = (rowStrideElems & 0xFFFFu) << 16;           // tensor_dim0[15:0] @ bits63:48
    g1.s2 = (rowStrideElems >> 16) | ((tensorRows & 0xFFFFu) << 16); // dim0 hi | dim1 lo
    g1.s3 = (tensorRows >> 16) | (cols << 16);          // dim1 hi | tile_dim0
    g1.s4 = rows;                                       // tile_dim1 | tile_dim2=0
    g1.s5 = rowStrideElems;                             // tensor_dim0_stride[31:0]
    g1.s6 = 0u;                                         // stride0 hi | stride1 lo
    g1.s7 = 0u;                                         // stride1 hi
    asm volatile("tensor_load_to_lds %0, %1" :: "s"(g0), "s"(g1) : "memory");
}
#define WAIT_TENSOR() __builtin_amdgcn_s_wait_tensorcnt(0)

template <bool BT>
__global__ void k_gemm_wmma(const unsigned short* __restrict__ A, long long sA,
                            const unsigned short* __restrict__ Bm, long long sB,
                            const float* __restrict__ bias,
                            const float* __restrict__ residual,
                            float* __restrict__ outF,
                            unsigned short* __restrict__ outH,
                            long long sC, int M, int N, int K, int flags, float scale) {
    __shared__ unsigned short As[2][64][32];   // [buf][m][k]
    __shared__ unsigned short Bs[2][32][64];   // [buf][k][n]

    const int tid  = threadIdx.x;
    const int lane = tid & 31;
    const int wave = tid >> 5;
    const int wy = wave >> 1, wx = wave & 1;        // 2x2 waves -> 64x64 block
    const int blockRow = blockIdx.y * 64;
    const int blockCol = blockIdx.x * 64;
    const int z = blockIdx.z;

    const unsigned short* Ab = A  + (size_t)z * sA;
    const unsigned short* Bb = Bm + (size_t)z * sB;

    v8f acc[2][2];
#pragma unroll
    for (int i = 0; i < 2; ++i)
#pragma unroll
        for (int j = 0; j < 2; ++j) acc[i][j] = (v8f){0.f, 0.f, 0.f, 0.f, 0.f, 0.f, 0.f, 0.f};

    union Frag { uint4 q[2]; v16bf v; };
    const int am   = wy * 32 + (lane & 15);
    const int koff = (lane < 16) ? 0 : 8;
    const int bn   = wx * 32;

    if (!BT) {
        // ---- async + TDM double-buffered pipeline (dims multiple of 64 here) ----
        auto issue = [&](int buf, int ks) {
            // A tile 64x32: per-lane async copies (ASYNCcnt)
            for (int c = tid; c < 256; c += 128) {
                int r = c >> 2, c8 = (c & 3) << 3;
                unsigned la = (unsigned)(size_t)&As[buf][r][c8];
                unsigned long long ga =
                    (unsigned long long)(size_t)(Ab + (size_t)(blockRow + r) * K + ks + c8);
                ASYNC_B128(la, ga);
            }
            // B tile 32x64: one TDM descriptor per workgroup (TENSORcnt)
            if (wave == 0) {
                unsigned la = (unsigned)(size_t)&Bs[buf][0][0];
                unsigned long long ga =
                    (unsigned long long)(size_t)(Bb + (size_t)ks * N + blockCol);
                tdm_load_tile_2d(la, ga, 64u, 32u, (unsigned)N, (unsigned)K);
            }
        };

        const int nk = K >> 5;
        issue(0, 0);
        WAIT_ASYNC();
        WAIT_TENSOR();
        __syncthreads();
        for (int t = 0; t < nk; ++t) {
            const int cur = t & 1;
            if (t + 1 < nk) issue(cur ^ 1, (t + 1) << 5);    // prefetch next tile

            Frag a[2], b[2];
#pragma unroll
            for (int mt = 0; mt < 2; ++mt) {
                a[mt].q[0] = *(const uint4*)&As[cur][am + mt * 16][koff];
                a[mt].q[1] = *(const uint4*)&As[cur][am + mt * 16][16 + koff];
            }
#pragma unroll
            for (int nt = 0; nt < 2; ++nt) {
#pragma unroll
                for (int s = 0; s < 16; ++s)
                    ((unsigned short*)&b[nt])[s] = Bs[cur][koff + s + (s & 8)][bn + nt * 16 + (lane & 15)];
            }
#pragma unroll
            for (int mt = 0; mt < 2; ++mt)
#pragma unroll
                for (int nt = 0; nt < 2; ++nt)
                    acc[mt][nt] = __builtin_amdgcn_wmma_f32_16x16x32_bf16(
                        false, a[mt].v, false, b[nt].v, (short)0, acc[mt][nt], false, false);

            WAIT_ASYNC();
            WAIT_TENSOR();
            __syncthreads();
        }
    } else {
        // ---- sync staging with transpose scatter (B is [N,K]; N may be ragged) ----
        for (int ks = 0; ks < K; ks += 32) {
            __syncthreads();
            for (int c = tid; c < 256; c += 128) {
                int r = c >> 2, c8 = (c & 3) << 3;
                *(uint4*)&As[0][r][c8] =
                    *(const uint4*)(Ab + (size_t)(blockRow + r) * K + ks + c8);
            }
            for (int c = tid; c < 256; c += 128) {
                int n = c >> 2, k8 = (c & 3) << 3;
                int ng = blockCol + n;
                if (ng < N) {
                    uint4 d = *(const uint4*)(Bb + (size_t)ng * K + ks + k8);
                    const unsigned short* ds = (const unsigned short*)&d;
#pragma unroll
                    for (int j = 0; j < 8; ++j) Bs[0][k8 + j][n] = ds[j];
                } else {
#pragma unroll
                    for (int j = 0; j < 8; ++j) Bs[0][k8 + j][n] = 0;
                }
            }
            __syncthreads();

            if (ks + 32 < K)
                __builtin_prefetch(Ab + (size_t)(blockRow + (tid >> 2)) * K + ks + 32, 0, 1);

            Frag a[2], b[2];
#pragma unroll
            for (int mt = 0; mt < 2; ++mt) {
                a[mt].q[0] = *(const uint4*)&As[0][am + mt * 16][koff];
                a[mt].q[1] = *(const uint4*)&As[0][am + mt * 16][16 + koff];
            }
#pragma unroll
            for (int nt = 0; nt < 2; ++nt) {
#pragma unroll
                for (int s = 0; s < 16; ++s)
                    ((unsigned short*)&b[nt])[s] = Bs[0][koff + s + (s & 8)][bn + nt * 16 + (lane & 15)];
            }
#pragma unroll
            for (int mt = 0; mt < 2; ++mt)
#pragma unroll
                for (int nt = 0; nt < 2; ++nt)
                    acc[mt][nt] = __builtin_amdgcn_wmma_f32_16x16x32_bf16(
                        false, a[mt].v, false, b[nt].v, (short)0, acc[mt][nt], false, false);
        }
    }

    // Epilogue. C 16x16 f32 layout: n = lane%16, m = vgpr + (lane>=16)*8
    const int lm = lane & 15, lh = lane >> 4;
#pragma unroll
    for (int mt = 0; mt < 2; ++mt) {
#pragma unroll
        for (int nt = 0; nt < 2; ++nt) {
            int ng = blockCol + wx * 32 + nt * 16 + lm;
            float bv = (bias && ng < N) ? bias[ng] : 0.f;
#pragma unroll
            for (int v = 0; v < 8; ++v) {
                int mg = blockRow + wy * 32 + mt * 16 + lh * 8 + v;
                float xv = acc[mt][nt][v] + bv;
                if (flags & FLAG_CAUSAL) { xv *= scale; if (ng > mg) xv = -1e30f; }
                if (ng < N) {
                    size_t idx = (size_t)z * sC + (size_t)mg * N + ng;
                    if (residual) xv += residual[idx];
                    if (flags & FLAG_GELU) {
                        float t = xv + 0.044715f * xv * xv * xv;
                        xv = 0.5f * xv * (1.f + tanhf(0.7978845608028654f * t));
                    }
                    if (outF) outF[idx] = xv;
                    if (outH) outH[idx] = f2bf(xv);
                }
            }
        }
    }
}

static void gemm(hipStream_t st, bool bt, const unsigned short* A, long long sA,
                 const unsigned short* B, long long sB, const float* bias,
                 const float* res, float* oF, unsigned short* oH, long long sC,
                 int M, int N, int K, int batch, int flags, float scale) {
    dim3 g((N + 63) / 64, M / 64, batch), b(128);
    if (bt) k_gemm_wmma<true ><<<g, b, 0, st>>>(A, sA, B, sB, bias, res, oF, oH, sC, M, N, K, flags, scale);
    else    k_gemm_wmma<false><<<g, b, 0, st>>>(A, sA, B, sB, bias, res, oF, oH, sC, M, N, K, flags, scale);
}

// ============================================================================
extern "C" void kernel_launch(void* const* d_in, const int* in_sizes, int n_in,
                              void* d_out, int out_size, void* d_ws, size_t ws_size,
                              hipStream_t stream) {
    (void)in_sizes; (void)n_in; (void)out_size; (void)ws_size;

    const int*   x      = (const int*)d_in[0];
    const int*   label  = (const int*)d_in[1];
    const float* wte    = (const float*)d_in[2];
    const float* wpe    = (const float*)d_in[3];
    const float* ln1_w  = (const float*)d_in[4];
    const float* ln1_b  = (const float*)d_in[5];
    const float* w_attn = (const float*)d_in[6];
    const float* b_attn = (const float*)d_in[7];
    const float* w_proj = (const float*)d_in[8];
    const float* b_proj = (const float*)d_in[9];
    const float* ln2_w  = (const float*)d_in[10];
    const float* ln2_b  = (const float*)d_in[11];
    const float* w_fc   = (const float*)d_in[12];
    const float* b_fc   = (const float*)d_in[13];
    const float* w_fcp  = (const float*)d_in[14];
    const float* b_fcp  = (const float*)d_in[15];
    const float* lnf_w  = (const float*)d_in[16];
    const float* lnf_b  = (const float*)d_in[17];

    const long long V = 50257, E = 768, T = 1024, BB = 2, L = 12, HH = 12, DH = 64;
    const long long M = BB * T, BH = BB * HH;

    // ---- workspace bump allocator (~460 MB) ----
    char* wsp = (char*)d_ws;
    auto alloc = [&](size_t bytes) -> void* {
        void* p = (void*)wsp;
        wsp += (bytes + 255) & ~(size_t)255;
        return p;
    };
    unsigned short* wte_h = (unsigned short*)alloc(V * E * 2);
    unsigned short* wat_h = (unsigned short*)alloc(L * E * 3 * E * 2);
    unsigned short* wpr_h = (unsigned short*)alloc(L * E * E * 2);
    unsigned short* wfc_h = (unsigned short*)alloc(L * E * 4 * E * 2);
    unsigned short* wfp_h = (unsigned short*)alloc(L * 4 * E * E * 2);
    float*          h     = (float*)alloc(M * E * 4);
    unsigned short* a_h   = (unsigned short*)alloc(M * E * 2);
    float*          qkv   = (float*)alloc(M * 3 * E * 4);
    unsigned short* q_h   = (unsigned short*)alloc(BH * T * DH * 2);
    unsigned short* kk_h  = (unsigned short*)alloc(BH * T * DH * 2);
    unsigned short* v_h   = (unsigned short*)alloc(BH * T * DH * 2);
    float*          S     = (float*)alloc(BH * T * T * 4);
    unsigned short* P_h   = (unsigned short*)alloc(BH * T * T * 2);
    float*          Y     = (float*)alloc(BH * T * DH * 4);
    unsigned short* y_h   = (unsigned short*)alloc(M * E * 2);
    unsigned short* m_h   = (unsigned short*)alloc(M * 4 * E * 2);

    float* logits = (float*)d_out;
    float* loss   = logits + (size_t)(M * V);

    // ---- convert all weights to bf16 (done every call; deterministic) ----
    k_f32_to_bf16<<<4096, 256, 0, stream>>>(wte,    wte_h, V * E);
    k_f32_to_bf16<<<4096, 256, 0, stream>>>(w_attn, wat_h, L * E * 3 * E);
    k_f32_to_bf16<<<4096, 256, 0, stream>>>(w_proj, wpr_h, L * E * E);
    k_f32_to_bf16<<<4096, 256, 0, stream>>>(w_fc,   wfc_h, L * E * 4 * E);
    k_f32_to_bf16<<<4096, 256, 0, stream>>>(w_fcp,  wfp_h, L * 4 * E * E);

    // ---- embeddings ----
    k_embed<<<(int)M, 256, 0, stream>>>(x, wte, wpe, h);

    const long long nBTE = M * E;
    const int packGrid = (int)((nBTE + 255) / 256);

    for (int l = 0; l < (int)L; ++l) {
        // ln1 -> a
        k_layernorm<<<(int)M, 256, 0, stream>>>(h, ln1_w + l * E, ln1_b + l * E, a_h);
        // qkv = a @ w_attn + b_attn   [2048 x 2304]
        gemm(stream, false, a_h, 0, wat_h + (size_t)l * E * 3 * E, 0,
             b_attn + (size_t)l * 3 * E, nullptr, qkv, nullptr, 0,
             (int)M, (int)(3 * E), (int)E, 1, 0, 1.f);
        k_pack_qkv<<<packGrid, 256, 0, stream>>>(qkv, q_h, kk_h, v_h);
        // S = (Q K^T) * 1/sqrt(D), causal masked   [24 x 1024 x 1024]
        gemm(stream, true, q_h, T * DH, kk_h, T * DH, nullptr, nullptr, S, nullptr,
             T * T, (int)T, (int)T, (int)DH, (int)BH, FLAG_CAUSAL, 0.125f);
        k_softmax_row<<<(int)(BH * T), 256, 0, stream>>>(S, P_h);
        // Y = P V   [24 x 1024 x 64]
        gemm(stream, false, P_h, T * T, v_h, T * DH, nullptr, nullptr, Y, nullptr,
             T * DH, (int)T, (int)DH, (int)T, (int)BH, 0, 1.f);
        k_pack_y<<<packGrid, 256, 0, stream>>>(Y, y_h);
        // h = h + y @ w_proj + b_proj
        gemm(stream, false, y_h, 0, wpr_h + (size_t)l * E * E, 0,
             b_proj + (size_t)l * E, h, h, nullptr, 0,
             (int)M, (int)E, (int)E, 1, 0, 1.f);
        // ln2 -> a
        k_layernorm<<<(int)M, 256, 0, stream>>>(h, ln2_w + l * E, ln2_b + l * E, a_h);
        // m = gelu(a @ w_fc + b_fc)  -> bf16 directly  [2048 x 3072]
        gemm(stream, false, a_h, 0, wfc_h + (size_t)l * E * 4 * E, 0,
             b_fc + (size_t)l * 4 * E, nullptr, nullptr, m_h, 0,
             (int)M, (int)(4 * E), (int)E, 1, FLAG_GELU, 1.f);
        // h = h + m @ w_fcp + b_fcp
        gemm(stream, false, m_h, 0, wfp_h + (size_t)l * 4 * E * E, 0,
             b_fcp + (size_t)l * E, h, h, nullptr, 0,
             (int)M, (int)E, (int)(4 * E), 1, 0, 1.f);
    }

    // final layernorm -> lm head (tied wte, transposed-B) -> logits in d_out
    k_layernorm<<<(int)M, 256, 0, stream>>>(h, lnf_w, lnf_b, a_h);
    gemm(stream, true, a_h, 0, wte_h, 0, nullptr, nullptr, logits, nullptr, 0,
         (int)M, (int)V, (int)E, 1, 0, 1.f);

    // loss = mean(logsumexp(row) - row[label])
    k_zero<<<1, 1, 0, stream>>>(loss);
    k_loss<<<(int)M, 256, 0, stream>>>(logits, label, loss);
}
